// PerformerAttention_5076651344514
// MI455X (gfx1250) — matrix operand order repaired and checked by reading, not compiled, not run on hardware
//
#include <hip/hip_runtime.h>
#include <hip/hip_bf16.h>

typedef __bf16 bf16_t;
typedef __attribute__((ext_vector_type(16))) __bf16 v16bf;
typedef __attribute__((ext_vector_type(8)))  __bf16 v8bf;
typedef __attribute__((ext_vector_type(8)))  float  v8f;
typedef __attribute__((ext_vector_type(4)))  unsigned int u32x4;
typedef __attribute__((ext_vector_type(4)))  int int4v;

#define B_  4
#define S_  4096
#define D_  1024
#define H_  16
#define DH_ 64
#define F_  256

enum { EP_BF16 = 0, EP_RELU_BF16 = 1, EP_BF16_T = 2, EP_F32_BIAS = 3, EP_DIV_BF16 = 4 };

#if defined(__HIP_DEVICE_COMPILE__) && __has_builtin(__builtin_amdgcn_global_load_async_to_lds_b128)
#define ASYNC_LDS 1
#else
#define ASYNC_LDS 0
#endif

#if ASYNC_LDS
typedef __attribute__((address_space(1))) int4v* gptr_b128;
typedef __attribute__((address_space(3))) int4v* lptr_b128;
__device__ __forceinline__ void async_copy_b128(const bf16_t* g, bf16_t* l) {
  __builtin_amdgcn_global_load_async_to_lds_b128((gptr_b128)g, (lptr_b128)l, 0, 0);
}
#endif

__device__ __forceinline__ void wait_async_lds() {
#if __has_builtin(__builtin_amdgcn_s_wait_asynccnt)
  __builtin_amdgcn_s_wait_asynccnt(0);
#else
  asm volatile("s_wait_asynccnt 0x0" ::: "memory");
#endif
}

// ---------------------------------------------------------------------------
// fp32 -> bf16 (grid-stride)
// ---------------------------------------------------------------------------
__global__ void f32_to_bf16_kernel(const float* __restrict__ s, bf16_t* __restrict__ d, long n) {
  long i = (long)blockIdx.x * blockDim.x + threadIdx.x;
  long stride = (long)gridDim.x * blockDim.x;
  for (; i < n; i += stride) d[i] = (bf16_t)s[i];
}

// s: [R,C] f32  ->  d: [C,R] bf16 (transpose + convert)
__global__ void transpose_cvt_kernel(const float* __restrict__ s, bf16_t* __restrict__ d,
                                     int R, int C) {
  long n = (long)R * C;
  long i = (long)blockIdx.x * blockDim.x + threadIdx.x;
  long stride = (long)gridDim.x * blockDim.x;
  for (; i < n; i += stride) {
    long r = i / C, c = i % C;
    d[c * (long)R + r] = (bf16_t)s[i];
  }
}

// ---------------------------------------------------------------------------
// Batched bf16 WMMA GEMM: C = op(A) @ op(B), 64x64 block tile, TK=32,
// 128 threads = 4 waves, each wave a 32x32 quadrant (2x2 WMMA tiles).
// LDS layout is always k-contiguous:  sA[m][k], sB[n][k]  (stride LK=40, padded).
//   TRA: global A is [K,M] (m contiguous)  -> transpose while staging
//   TRB: global B is [K,N] (n contiguous)  -> transpose while staging
//  !TRA: global A is [M,K];  !TRB: global B is [N,K] (pre-transposed weights)
// Non-transposing copies use async global->LDS; double-buffered pipeline.
// ---------------------------------------------------------------------------
template<bool TRA, bool TRB>
__global__ __launch_bounds__(128)
void gemm_bf16_kernel(const bf16_t* __restrict__ A, const bf16_t* __restrict__ Bm,
                      const float* __restrict__ bias, const float* __restrict__ norm,
                      float* __restrict__ Cf, bf16_t* __restrict__ Cb,
                      int M, int N, int K, int lda, int ldb, int ldc,
                      long aOuter, long aInner, long bOuter, long bInner,
                      long cOuter, long cInner, int innerCount, long normStride,
                      int mode)
{
  constexpr int LK = 40; // padded LDS row stride (elements); 80B keeps 16B alignment
  __shared__ __attribute__((aligned(16))) bf16_t sA[2][64 * LK];
  __shared__ __attribute__((aligned(16))) bf16_t sB[2][64 * LK];

  const int z  = blockIdx.z;
  const int zo = z / innerCount, zi = z % innerCount;
  const bf16_t* Abase = A  + (long)zo * aOuter + (long)zi * aInner;
  const bf16_t* Bbase = Bm + (long)zo * bOuter + (long)zi * bInner;
  const long coff = (long)zo * cOuter + (long)zi * cInner;

  const int m0 = blockIdx.y * 64;
  const int n0 = blockIdx.x * 64;
  const int t    = threadIdx.x;
  const int lane = t & 31;
  const int wave = t >> 5;
  const int wm   = (wave >> 1) * 32;
  const int wn   = (wave &  1) * 32;
  const int l15  = lane & 15;
  const int half = lane >> 4;

  v8f acc[2][2] = {};

  auto stage = [&](int k0, int p) {
#pragma unroll
    for (int i = 0; i < 2; ++i) {
      int c = t + i * 128;
      // ---- A tile ----
      if (TRA) { // global rows = k (32), contiguous along m
        int kr = c >> 3, mc = (c & 7) * 8;
        v8bf g = *reinterpret_cast<const v8bf*>(Abase + (long)(k0 + kr) * lda + m0 + mc);
#pragma unroll
        for (int j = 0; j < 8; ++j) sA[p][(mc + j) * LK + kr] = g[j];
      } else {   // global rows = m (64), contiguous along k
        int mr = c >> 2, kc = (c & 3) * 8;
        const bf16_t* src = Abase + (long)(m0 + mr) * lda + k0 + kc;
        bf16_t* dst = &sA[p][mr * LK + kc];
#if ASYNC_LDS
        async_copy_b128(src, dst);
#else
        *reinterpret_cast<u32x4*>(dst) = *reinterpret_cast<const u32x4*>(src);
#endif
      }
      // ---- B tile ----
      if (TRB) { // global rows = k (32), contiguous along n
        int kr = c >> 3, nc = (c & 7) * 8;
        v8bf g = *reinterpret_cast<const v8bf*>(Bbase + (long)(k0 + kr) * ldb + n0 + nc);
#pragma unroll
        for (int j = 0; j < 8; ++j) sB[p][(nc + j) * LK + kr] = g[j];
      } else {   // global rows = n (64), contiguous along k
        int nr = c >> 2, kc = (c & 3) * 8;
        const bf16_t* src = Bbase + (long)(n0 + nr) * ldb + k0 + kc;
        bf16_t* dst = &sB[p][nr * LK + kc];
#if ASYNC_LDS
        async_copy_b128(src, dst);
#else
        *reinterpret_cast<u32x4*>(dst) = *reinterpret_cast<const u32x4*>(src);
#endif
      }
    }
  };

  stage(0, 0);
#if ASYNC_LDS
  if (!TRA || !TRB) wait_async_lds();
#endif
  __syncthreads();

  int p = 0;
  for (int k0 = 0; k0 < K; k0 += 32) {
    if (k0 + 32 < K) stage(k0 + 32, p ^ 1); // prefetch next slab into other buffer

    // ---- fragment gather: all k-contiguous -> 2x ds_load_b128 per fragment ----
    const bf16_t* a = &sA[p][0];
    const bf16_t* b = &sB[p][0];
    v16bf af[2], bfr[2];
#pragma unroll
    for (int mi = 0; mi < 2; ++mi) {
      int m = wm + mi * 16 + l15;
      v8bf lo = *reinterpret_cast<const v8bf*>(&a[m * LK + half * 8]);      // k = half*8 + 0..7
      v8bf hi = *reinterpret_cast<const v8bf*>(&a[m * LK + 16 + half * 8]); // k = 16 + half*8 + 0..7
      af[mi] = __builtin_shufflevector(lo, hi, 0,1,2,3,4,5,6,7,8,9,10,11,12,13,14,15);
    }
#pragma unroll
    for (int ni = 0; ni < 2; ++ni) {
      int n = wn + ni * 16 + l15;
      v8bf lo = *reinterpret_cast<const v8bf*>(&b[n * LK + half * 8]);      // k = half*8 + 0..7
      v8bf hi = *reinterpret_cast<const v8bf*>(&b[n * LK + 16 + half * 8]); // k = 16 + half*8 + 0..7
      bfr[ni] = __builtin_shufflevector(lo, hi, 0,1,2,3,4,5,6,7,8,9,10,11,12,13,14,15);
    }
#pragma unroll
    for (int mi = 0; mi < 2; ++mi)
#pragma unroll
      for (int ni = 0; ni < 2; ++ni)
        acc[mi][ni] = __builtin_amdgcn_wmma_f32_16x16x32_bf16(
            false, af[mi], false, bfr[ni], (short)0, acc[mi][ni], false, false);

#if ASYNC_LDS
    if (!TRA || !TRB) wait_async_lds();
#endif
    __syncthreads();
    p ^= 1;
  }

  // ---- epilogue: lane holds (m = half*8 + r, n = l15) per 16x16 tile ----
#pragma unroll
  for (int mi = 0; mi < 2; ++mi) {
#pragma unroll
    for (int ni = 0; ni < 2; ++ni) {
#pragma unroll
      for (int r = 0; r < 8; ++r) {
        int  m  = wm + mi * 16 + half * 8 + r;
        int  n  = wn + ni * 16 + l15;
        long gm = (long)m0 + m;
        long gn = (long)n0 + n;
        float v = acc[mi][ni][r];
        switch (mode) {
          case EP_BF16:      Cb[coff + gm * ldc + gn] = (bf16_t)v; break;
          case EP_RELU_BF16: Cb[coff + gm * ldc + gn] = (bf16_t)fmaxf(v, 0.f); break;
          case EP_BF16_T:    Cb[coff + gn * ldc + gm] = (bf16_t)v; break;
          case EP_F32_BIAS:  Cf[coff + gm * ldc + gn] = v + bias[gn]; break;
          case EP_DIV_BF16: {
            float nv = norm[(long)z * normStride + gm];
            Cb[coff + gm * ldc + gn] = (bf16_t)(v / (nv + 1e-8f));
          } break;
        }
      }
    }
  }
}

// ---------------------------------------------------------------------------
// ksum[bh][f] = sum_s Kp[bh][s][f]
// ---------------------------------------------------------------------------
__global__ __launch_bounds__(256)
void colsum_kernel(const bf16_t* __restrict__ kp, float* __restrict__ ksum) {
  long bh = blockIdx.x;
  int  f  = threadIdx.x;
  const bf16_t* p = kp + bh * (long)S_ * F_ + f;
  float acc = 0.f;
  for (int s = 0; s < S_; ++s) acc += (float)p[(long)s * F_];
  ksum[bh * F_ + f] = acc;
}

// ---------------------------------------------------------------------------
// norm[bh*S+s] = dot(Qp[bh][s][:], ksum[bh][:])  (one wave32 per row)
// ---------------------------------------------------------------------------
__global__ __launch_bounds__(256)
void rownorm_kernel(const bf16_t* __restrict__ qp, const float* __restrict__ ksum,
                    float* __restrict__ norm) {
  int  wave = threadIdx.x >> 5, lane = threadIdx.x & 31;
  long row = (long)blockIdx.x * 8 + wave;
  long bh  = row >> 12; // / S_
  const bf16_t* q  = qp + row * F_;
  const float*  ks = ksum + bh * F_;
  float acc = 0.f;
#pragma unroll
  for (int j = 0; j < F_ / 32; ++j) {
    int f = lane + j * 32;
    acc += (float)q[f] * ks[f];
  }
#pragma unroll
  for (int off = 16; off > 0; off >>= 1) acc += __shfl_xor(acc, off, 32);
  if (lane == 0) norm[row] = acc;
}

// ---------------------------------------------------------------------------
extern "C" void kernel_launch(void* const* d_in, const int* in_sizes, int n_in,
                              void* d_out, int out_size, void* d_ws, size_t ws_size,
                              hipStream_t stream) {
  const float* hs   = (const float*)d_in[0];
  const float* Wq   = (const float*)d_in[1];
  const float* Wk   = (const float*)d_in[2];
  const float* Wv   = (const float*)d_in[3];
  const float* proj = (const float*)d_in[4];
  const float* Wo   = (const float*)d_in[5];
  const float* bo   = (const float*)d_in[6];
  float* out = (float*)d_out;
  (void)in_sizes; (void)n_in; (void)out_size; (void)ws_size;

  char* ws = (char*)d_ws;
  size_t off = 0;
  auto take = [&](size_t bytes) -> char* {
    char* p = ws + off;
    off = (off + bytes + 255) & ~(size_t)255;
    return p;
  };
  const long M0 = (long)B_ * S_;            // 16384
  const long SD = (long)S_ * D_, SF = (long)S_ * F_, FD = (long)F_ * DH_;

  bf16_t* Xb   = (bf16_t*)take(M0 * D_ * 2);
  bf16_t* Wqt  = (bf16_t*)take((long)D_ * D_ * 2);   // [N,K] = Wq^T
  bf16_t* Wkt  = (bf16_t*)take((long)D_ * D_ * 2);
  bf16_t* Wvt  = (bf16_t*)take((long)D_ * D_ * 2);
  bf16_t* Pt   = (bf16_t*)take((long)F_ * DH_ * 2);  // [F,DH] = proj^T
  bf16_t* Wot  = (bf16_t*)take((long)D_ * D_ * 2);
  bf16_t* Qb   = (bf16_t*)take(M0 * D_ * 2);
  bf16_t* Kb   = (bf16_t*)take(M0 * D_ * 2);
  bf16_t* Vb   = (bf16_t*)take(M0 * D_ * 2);
  bf16_t* Qp   = (bf16_t*)take((long)B_ * H_ * S_ * F_ * 2);
  bf16_t* Kp   = (bf16_t*)take((long)B_ * H_ * S_ * F_ * 2);
  bf16_t* KVt  = (bf16_t*)take((long)B_ * H_ * DH_ * F_ * 2); // [DH,F] per head
  float*  Ksum = (float*)take((long)B_ * H_ * F_ * 4);
  float*  Nrm  = (float*)take((long)B_ * H_ * S_ * 4);
  bf16_t* Attb = (bf16_t*)take(M0 * D_ * 2);

  f32_to_bf16_kernel<<<dim3(2048), dim3(256), 0, stream>>>(hs, Xb, M0 * D_);
  transpose_cvt_kernel<<<dim3(1024), dim3(256), 0, stream>>>(Wq, Wqt, D_, D_);
  transpose_cvt_kernel<<<dim3(1024), dim3(256), 0, stream>>>(Wk, Wkt, D_, D_);
  transpose_cvt_kernel<<<dim3(1024), dim3(256), 0, stream>>>(Wv, Wvt, D_, D_);
  transpose_cvt_kernel<<<dim3(64),   dim3(256), 0, stream>>>(proj, Pt, DH_, F_);
  transpose_cvt_kernel<<<dim3(1024), dim3(256), 0, stream>>>(Wo, Wot, D_, D_);

  auto gemm = [&](bool trAB, const bf16_t* A, const bf16_t* Bm,
                  const float* bias, const float* norm,
                  float* Cf, bf16_t* Cb,
                  int M, int N, int K, int lda, int ldb, int ldc,
                  long aOuter, long aInner, long bOuter, long bInner,
                  long cOuter, long cInner, int innerCount, long normStride,
                  int mode, int batch) {
    dim3 grid(N / 64, M / 64, batch), block(128);
    if (trAB)
      gemm_bf16_kernel<true, true><<<grid, block, 0, stream>>>(
          A, Bm, bias, norm, Cf, Cb, M, N, K, lda, ldb, ldc,
          aOuter, aInner, bOuter, bInner, cOuter, cInner, innerCount, normStride, mode);
    else
      gemm_bf16_kernel<false, false><<<grid, block, 0, stream>>>(
          A, Bm, bias, norm, Cf, Cb, M, N, K, lda, ldb, ldc,
          aOuter, aInner, bOuter, bInner, cOuter, cInner, innerCount, normStride, mode);
  };

  // 1) Q/K/V = X @ W*   (B pre-transposed, ldb = K = D)
  gemm(false, Xb, Wqt, nullptr, nullptr, nullptr, Qb, (int)M0, D_, D_, D_, D_, D_,
       0, 0, 0, 0, 0, 0, 1, 0, EP_BF16, 1);
  gemm(false, Xb, Wkt, nullptr, nullptr, nullptr, Kb, (int)M0, D_, D_, D_, D_, D_,
       0, 0, 0, 0, 0, 0, 1, 0, EP_BF16, 1);
  gemm(false, Xb, Wvt, nullptr, nullptr, nullptr, Vb, (int)M0, D_, D_, D_, D_, D_,
       0, 0, 0, 0, 0, 0, 1, 0, EP_BF16, 1);

  // 2) q' = relu(Q_head @ proj), k' = relu(K_head @ proj); batch = B*H head slices
  gemm(false, Qb, Pt, nullptr, nullptr, nullptr, Qp, S_, F_, DH_, D_, DH_, F_,
       SD, DH_, 0, 0, (long)H_ * SF, SF, H_, 0, EP_RELU_BF16, B_ * H_);
  gemm(false, Kb, Pt, nullptr, nullptr, nullptr, Kp, S_, F_, DH_, D_, DH_, F_,
       SD, DH_, 0, 0, (long)H_ * SF, SF, H_, 0, EP_RELU_BF16, B_ * H_);

  // 3) kv = k'^T @ V_head : [F,DH]; store transposed -> KVt [DH,F] per head
  gemm(true, Kp, Vb, nullptr, nullptr, nullptr, KVt, F_, DH_, S_, F_, D_, F_,
       (long)H_ * SF, SF, SD, DH_, (long)H_ * FD, FD, H_, 0, EP_BF16_T, B_ * H_);

  // 4) ksum and normalizer
  colsum_kernel<<<dim3(B_ * H_), dim3(256), 0, stream>>>(Kp, Ksum);
  rownorm_kernel<<<dim3((B_ * H_ * S_) / 8), dim3(256), 0, stream>>>(Qp, Ksum, Nrm);

  // 5) attn = (q' @ kv) / (norm + eps), merged-head bf16 [B,S,D]; B = KVt [DH,F]
  gemm(false, Qp, KVt, nullptr, Nrm, nullptr, Attb, S_, DH_, F_, F_, F_, D_,
       (long)H_ * SF, SF, (long)H_ * FD, FD, SD, DH_, H_, S_, EP_DIV_BF16, B_ * H_);

  // 6) out = attn @ Wo + bo  -> f32 [16384,1024]
  gemm(false, Attb, Wot, bo, nullptr, out, nullptr, (int)M0, D_, D_, D_, D_, D_,
       0, 0, 0, 0, 0, 0, 1, 0, EP_F32_BIAS, 1);
}
